// Block_37864431682616
// MI455X (gfx1250) — matrix operand order repaired and checked by reading, not compiled, not run on hardware
//
#include <hip/hip_runtime.h>
#include <stdint.h>

// ---- problem constants (match reference) ----
#define BQ    4
#define TT    1024
#define CCH   1024
#define NHH   16
#define NKVV  4
#define HDD   64
#define EE    8
#define FFH   4096
#define WINN  256
#define NTOK  (BQ*TT)   // 4096

typedef __attribute__((ext_vector_type(16))) __bf16 v16bf;
typedef __attribute__((ext_vector_type(8)))  float  v8f;
typedef __attribute__((ext_vector_type(4)))  unsigned int v4u;

__device__ __forceinline__ uint16_t f2bf(float f) {
  union { float f; uint32_t u; } c; c.f = f;
  uint32_t u = c.u;
  uint32_t r = u + 0x7FFFu + ((u >> 16) & 1u);   // round-to-nearest-even
  return (uint16_t)(r >> 16);
}

// WGP-scope prefetch (scope 0 = pull into all cache levels incl. WGP$).
// __builtin_prefetch can only reach SE/SYS scopes, which skip the near cache.
__device__ __forceinline__ void prefetch_wgp_1k(const void* p) {
  asm volatile("global_prefetch_b8 %0, off offset:1024" :: "v"(p));
}

// ---------------- RMSNorm: fp32 in -> bf16 (and optional fp32) out ----------------
__global__ void rmsnorm_kernel(const float* __restrict__ x, const float* __restrict__ w,
                               uint16_t* __restrict__ obf, float* __restrict__ of32) {
  int row = blockIdx.x;
  const float* xr = x + (size_t)row * CCH;
  __shared__ float red[256];
  float s = 0.f;
  for (int i = threadIdx.x; i < CCH; i += 256) { float v = xr[i]; s += v * v; }
  red[threadIdx.x] = s;
  __syncthreads();
  for (int off = 128; off > 0; off >>= 1) {
    if (threadIdx.x < off) red[threadIdx.x] += red[threadIdx.x + off];
    __syncthreads();
  }
  float scale = rsqrtf(red[0] / (float)CCH + 1e-6f);
  for (int i = threadIdx.x; i < CCH; i += 256) {
    float v = xr[i] * scale * w[i];
    obf[(size_t)row * CCH + i] = f2bf(v);
    if (of32) of32[(size_t)row * CCH + i] = v;
  }
}

// ---------------- transpose + fp32->bf16 convert: in[K,N] -> out[N,K] ----------------
__global__ void convT_kernel(const float* __restrict__ in, uint16_t* __restrict__ out,
                             int K, int N) {
  __shared__ float tile[32][33];
  int k0 = blockIdx.x * 32, n0 = blockIdx.y * 32;
  int tx = threadIdx.x, ty = threadIdx.y;           // block (32,8)
  #pragma unroll
  for (int i = 0; i < 4; ++i)
    tile[ty + 8 * i][tx] = in[(size_t)(k0 + ty + 8 * i) * N + n0 + tx];
  __syncthreads();
  #pragma unroll
  for (int i = 0; i < 4; ++i)
    out[(size_t)(n0 + ty + 8 * i) * K + k0 + tx] = f2bf(tile[tx][ty + 8 * i]);
}

// ---------------- bf16 WMMA GEMM: C[M,N] = A[M,K] * Bt[N,K]^T ----------------
// Wave tile 32(M)x64(N); block = 8 waves as 4(M)x2(N) -> 128x128 block tile.
// Register double-buffered over K (step 64, two 32-wide stages).
// mode 0: C = acc     mode 1: C = acc + aux[m,n]     mode 2: C += aux[m*auxStride]*acc
union ABfrag { v16bf v; v4u u[2]; };

__device__ __forceinline__ void load_frags(const uint16_t* __restrict__ aRow0,
                                           const uint16_t* __restrict__ aRow1,
                                           const uint16_t* const (&bRow)[4], int k0,
                                           ABfrag (&aF)[2], ABfrag (&bF)[4]) {
  aF[0].u[0] = *(const v4u*)(aRow0 + k0);
  aF[0].u[1] = *(const v4u*)(aRow0 + k0 + 16);
  aF[1].u[0] = *(const v4u*)(aRow1 + k0);
  aF[1].u[1] = *(const v4u*)(aRow1 + k0 + 16);
  #pragma unroll
  for (int j = 0; j < 4; ++j) {
    bF[j].u[0] = *(const v4u*)(bRow[j] + k0);
    bF[j].u[1] = *(const v4u*)(bRow[j] + k0 + 16);
  }
}

__device__ __forceinline__ void do_wmma(ABfrag (&aF)[2], ABfrag (&bF)[4], v8f (&acc)[2][4]) {
  #pragma unroll
  for (int i = 0; i < 2; ++i)
    #pragma unroll
    for (int j = 0; j < 4; ++j)
      acc[i][j] = __builtin_amdgcn_wmma_f32_16x16x32_bf16(
          false, aF[i].v, false, bF[j].v, (short)0, acc[i][j], false, false);
}

__global__ void __launch_bounds__(256)
wmma_gemm_bf16(const uint16_t* __restrict__ A, const uint16_t* __restrict__ Bt,
               float* __restrict__ C, int M, int N, int K,
               int mode, const float* __restrict__ aux, int auxStride) {
  int wave = threadIdx.x >> 5;
  int lane = threadIdx.x & 31;
  int m16  = lane & 15;
  int half = lane >> 4;
  int wm = wave >> 1, wn = wave & 1;
  int tileM = blockIdx.x * 128 + wm * 32;
  int tileN = blockIdx.y * 128 + wn * 64;

  v8f acc[2][4];
  #pragma unroll
  for (int i = 0; i < 2; ++i)
    #pragma unroll
    for (int j = 0; j < 4; ++j)
      #pragma unroll
      for (int r = 0; r < 8; ++r) acc[i][j][r] = 0.f;

  // A fragment: lane(m16,half) -> K offsets {8h..8h+7} and {16+8h..+7} (folded +8*half)
  const uint16_t* aRow0 = A + (size_t)(tileM + m16) * K + 8 * half;
  const uint16_t* aRow1 = aRow0 + (size_t)16 * K;
  // B fragment: lane(n=m16,half) -> K offsets {8h..8h+7} and {16+8h..+7} (folded +8*half)
  const uint16_t* bRow[4];
  #pragma unroll
  for (int j = 0; j < 4; ++j)
    bRow[j] = Bt + (size_t)(tileN + j * 16 + m16) * K + 8 * half;

  ABfrag aX[2], bX[4], aY[2], bY[4];
  load_frags(aRow0, aRow1, bRow, 0, aX, bX);

  // K assumed multiple of 64 (1024 / 4096 in this block)
  for (int k0 = 0; k0 < K; k0 += 64) {
    load_frags(aRow0, aRow1, bRow, k0 + 32, aY, bY);
    // WGP-scope prefetch 1KB (16 K-steps) ahead on each stream (fills near cache)
    prefetch_wgp_1k(aRow0 + k0);
    prefetch_wgp_1k(aRow1 + k0);
    prefetch_wgp_1k(bRow[0] + k0);
    prefetch_wgp_1k(bRow[1] + k0);
    prefetch_wgp_1k(bRow[2] + k0);
    prefetch_wgp_1k(bRow[3] + k0);
    do_wmma(aX, bX, acc);
    if (k0 + 64 < K)
      load_frags(aRow0, aRow1, bRow, k0 + 64, aX, bX);
    do_wmma(aY, bY, acc);
  }

  #pragma unroll
  for (int i = 0; i < 2; ++i) {
    #pragma unroll
    for (int j = 0; j < 4; ++j) {
      int col = tileN + j * 16 + m16;
      #pragma unroll
      for (int r = 0; r < 8; ++r) {
        int row = tileM + i * 16 + r + 8 * half;    // D layout: VGPR r -> M = r + 8*half
        size_t idx = (size_t)row * N + col;
        float vacc = acc[i][j][r];
        if (mode == 0)      C[idx] = vacc;
        else if (mode == 1) C[idx] = vacc + aux[idx];
        else                C[idx] += aux[(size_t)row * auxStride] * vacc;
      }
    }
  }
}

// ---------------- RoPE (in place, fp32) ----------------
__global__ void rope_kernel(float* __restrict__ t, int heads, int cols, int total) {
  int id = blockIdx.x * 256 + threadIdx.x;
  if (id >= total) return;
  int i   = id & 31;
  int hh  = (id >> 5) % heads;
  int row = id / (32 * heads);
  int pos = row % TT;
  float freq = __expf(-(float)(2 * i) * (logf(10000.f) / (float)HDD));
  float ang = (float)pos * freq;
  float sn, cs;
  __sincosf(ang, &sn, &cs);
  size_t base = (size_t)row * cols + hh * HDD;
  float x1 = t[base + i], x2 = t[base + i + 32];
  t[base + i]      = x1 * cs - x2 * sn;
  t[base + i + 32] = x2 * cs + x1 * sn;
}

// ---------------- sliding-window GQA attention (fp32 VALU), bf16 out ----------------
__global__ void __launch_bounds__(256)
attn_kernel(const float* __restrict__ q, const float* __restrict__ k,
            const float* __restrict__ v, uint16_t* __restrict__ obf) {
  __shared__ float qs[8][HDD];
  __shared__ float sc[8][WINN];
  int wave = threadIdx.x >> 5, lane = threadIdx.x & 31;
  int bh = blockIdx.y;
  int b = bh / NHH, h = bh % NHH, kvh = h / (NHH / NKVV);
  int qi = blockIdx.x * 8 + wave;

  size_t qoff = (size_t)(b * TT + qi) * CCH + h * HDD;
  qs[wave][2 * lane]     = q[qoff + 2 * lane]     * 0.125f;   // 1/sqrt(64)
  qs[wave][2 * lane + 1] = q[qoff + 2 * lane + 1] * 0.125f;
  __syncthreads();

  int jstart = qi - (WINN - 1); if (jstart < 0) jstart = 0;
  int nj = qi - jstart + 1;

  float lmax = -3.4e38f;
  for (int jj = lane; jj < nj; jj += 32) {
    const float* kr = k + (size_t)(b * TT + jstart + jj) * (NKVV * HDD) + kvh * HDD;
    float s = 0.f;
    #pragma unroll
    for (int d = 0; d < HDD; ++d) s += qs[wave][d] * kr[d];
    sc[wave][jj] = s;
    if (s > lmax) lmax = s;
  }
  #pragma unroll
  for (int m = 16; m > 0; m >>= 1) {
    float o = __shfl_xor(lmax, m, 32);
    lmax = lmax > o ? lmax : o;
  }
  __syncthreads();

  float lsum = 0.f;
  for (int jj = lane; jj < nj; jj += 32) {
    float e = __expf(sc[wave][jj] - lmax);
    sc[wave][jj] = e;
    lsum += e;
  }
  #pragma unroll
  for (int m = 16; m > 0; m >>= 1) lsum += __shfl_xor(lsum, m, 32);
  __syncthreads();

  float rcp = 1.f / lsum;
  int d0 = 2 * lane;
  float o0 = 0.f, o1 = 0.f;
  for (int jj = 0; jj < nj; ++jj) {
    float p = sc[wave][jj];
    const float* vr = v + (size_t)(b * TT + jstart + jj) * (NKVV * HDD) + kvh * HDD + d0;
    o0 += p * vr[0];
    o1 += p * vr[1];
  }
  size_t ooff = (size_t)(b * TT + qi) * CCH + h * HDD + d0;
  obf[ooff]     = f2bf(o0 * rcp);
  obf[ooff + 1] = f2bf(o1 * rcp);
}

// ---------------- gate logits: g[N,C] @ gw[C,E] ----------------
__global__ void gate_logits_kernel(const float* __restrict__ g, const float* __restrict__ gw,
                                   float* __restrict__ logits) {
  int id = blockIdx.x * 256 + threadIdx.x;
  if (id >= NTOK * EE) return;
  int n = id / EE, e = id % EE;
  const float* gr = g + (size_t)n * CCH;
  float s = 0.f;
  for (int c = 0; c < CCH; ++c) s += gr[c] * gw[(size_t)c * EE + e];
  logits[id] = s;
}

// ---------------- softmax + top-2 + renorm -> comb[N,E] ----------------
__global__ void topk_kernel(const float* __restrict__ logits, float* __restrict__ comb) {
  int n = blockIdx.x * 256 + threadIdx.x;
  if (n >= NTOK) return;
  float l[EE], p[EE];
  float m = -3.4e38f;
  for (int e = 0; e < EE; ++e) { l[e] = logits[n * EE + e]; if (l[e] > m) m = l[e]; }
  float s = 0.f;
  for (int e = 0; e < EE; ++e) { p[e] = __expf(l[e] - m); s += p[e]; }
  for (int e = 0; e < EE; ++e) p[e] /= s;
  int i1 = 0;
  for (int e = 1; e < EE; ++e) if (p[e] > p[i1]) i1 = e;
  int i2 = (i1 == 0) ? 1 : 0;
  for (int e = 0; e < EE; ++e) if (e != i1 && p[e] > p[i2]) i2 = e;
  float w = p[i1] + p[i2];
  for (int e = 0; e < EE; ++e) comb[n * EE + e] = 0.f;
  comb[n * EE + i1] = p[i1] / w;
  comb[n * EE + i2] = p[i2] / w;
}

// ---------------- silu(up1)*up3 -> bf16 ----------------
__global__ void silu_mul_kernel(const float* __restrict__ a, const float* __restrict__ b,
                                uint16_t* __restrict__ o, int total) {
  int id = blockIdx.x * 256 + threadIdx.x;
  if (id >= total) return;
  float x = a[id];
  float sl = x / (1.f + __expf(-x));
  o[id] = f2bf(sl * b[id]);
}

// ---------------- vectorized fp32 copy (d_out = h) ----------------
__global__ void copy_kernel(const float* __restrict__ src, float* __restrict__ dst, int total4) {
  int id = blockIdx.x * 256 + threadIdx.x;
  if (id < total4) ((float4*)dst)[id] = ((const float4*)src)[id];
}

// =============================== launcher ===============================
extern "C" void kernel_launch(void* const* d_in, const int* in_sizes, int n_in,
                              void* d_out, int out_size, void* d_ws, size_t ws_size,
                              hipStream_t stream) {
  (void)in_sizes; (void)n_in; (void)out_size; (void)ws_size;
  const float* x   = (const float*)d_in[0];
  const float* anw = (const float*)d_in[1];
  const float* fnw = (const float*)d_in[2];
  const float* wq  = (const float*)d_in[3];
  const float* wk  = (const float*)d_in[4];
  const float* wv  = (const float*)d_in[5];
  const float* wo  = (const float*)d_in[6];
  const float* gw  = (const float*)d_in[7];
  const float* w1  = (const float*)d_in[8];
  const float* w2  = (const float*)d_in[9];
  const float* w3  = (const float*)d_in[10];
  float* out = (float*)d_out;

  char* ws = (char*)d_ws;
  size_t off = 0;
  auto alloc = [&](size_t bytes) -> void* {
    void* p = ws + off;
    off = (off + bytes + 255) & ~(size_t)255;
    return p;
  };

  uint16_t* wqT   = (uint16_t*)alloc((size_t)CCH * CCH * 2);          // [C, C] -> [N=C, K=C]
  uint16_t* wkT   = (uint16_t*)alloc((size_t)(NKVV*HDD) * CCH * 2);   // [256, 1024]
  uint16_t* wvT   = (uint16_t*)alloc((size_t)(NKVV*HDD) * CCH * 2);
  uint16_t* woT   = (uint16_t*)alloc((size_t)CCH * CCH * 2);
  uint16_t* weT1  = (uint16_t*)alloc((size_t)FFH * CCH * 2);          // per-expert, reused
  uint16_t* weT3  = (uint16_t*)alloc((size_t)FFH * CCH * 2);
  uint16_t* weT2  = (uint16_t*)alloc((size_t)CCH * FFH * 2);
  uint16_t* hb    = (uint16_t*)alloc((size_t)NTOK * CCH * 2);         // rmsnorm(x) bf16
  float*    qbuf  = (float*)   alloc((size_t)NTOK * CCH * 4);
  float*    kbuf  = (float*)   alloc((size_t)NTOK * (NKVV*HDD) * 4);
  float*    vbuf  = (float*)   alloc((size_t)NTOK * (NKVV*HDD) * 4);
  uint16_t* attnb = (uint16_t*)alloc((size_t)NTOK * CCH * 2);         // attn out bf16
  float*    hbuf  = (float*)   alloc((size_t)NTOK * CCH * 4);         // h = x + y
  uint16_t* gbf   = (uint16_t*)alloc((size_t)NTOK * CCH * 2);         // rmsnorm(h) bf16
  float*    gf32  = (float*)   alloc((size_t)NTOK * CCH * 4);         // rmsnorm(h) fp32 (gate)
  float*    logit = (float*)   alloc((size_t)NTOK * EE * 4);
  float*    comb  = (float*)   alloc((size_t)NTOK * EE * 4);
  float*    up1   = (float*)   alloc((size_t)NTOK * FFH * 4);
  float*    up3   = (float*)   alloc((size_t)NTOK * FFH * 4);
  uint16_t* hidb  = (uint16_t*)alloc((size_t)NTOK * FFH * 2);

  dim3 cb(32, 8);

  // ---- attention sub-block ----
  rmsnorm_kernel<<<NTOK, 256, 0, stream>>>(x, anw, hb, nullptr);

  convT_kernel<<<dim3(CCH/32, CCH/32), cb, 0, stream>>>(wq, wqT, CCH, CCH);
  convT_kernel<<<dim3(CCH/32, (NKVV*HDD)/32), cb, 0, stream>>>(wk, wkT, CCH, NKVV*HDD);
  convT_kernel<<<dim3(CCH/32, (NKVV*HDD)/32), cb, 0, stream>>>(wv, wvT, CCH, NKVV*HDD);
  convT_kernel<<<dim3(CCH/32, CCH/32), cb, 0, stream>>>(wo, woT, CCH, CCH);

  wmma_gemm_bf16<<<dim3(NTOK/128, CCH/128), 256, 0, stream>>>(
      hb, wqT, qbuf, NTOK, CCH, CCH, 0, nullptr, 0);
  wmma_gemm_bf16<<<dim3(NTOK/128, (NKVV*HDD)/128), 256, 0, stream>>>(
      hb, wkT, kbuf, NTOK, NKVV*HDD, CCH, 0, nullptr, 0);
  wmma_gemm_bf16<<<dim3(NTOK/128, (NKVV*HDD)/128), 256, 0, stream>>>(
      hb, wvT, vbuf, NTOK, NKVV*HDD, CCH, 0, nullptr, 0);

  int qtot = NTOK * NHH * 32;
  rope_kernel<<<(qtot + 255)/256, 256, 0, stream>>>(qbuf, NHH, CCH, qtot);
  int ktot = NTOK * NKVV * 32;
  rope_kernel<<<(ktot + 255)/256, 256, 0, stream>>>(kbuf, NKVV, NKVV*HDD, ktot);

  attn_kernel<<<dim3(TT/8, BQ*NHH), 256, 0, stream>>>(qbuf, kbuf, vbuf, attnb);

  // y = attn @ wo ; h = x + y
  wmma_gemm_bf16<<<dim3(NTOK/128, CCH/128), 256, 0, stream>>>(
      attnb, woT, hbuf, NTOK, CCH, CCH, 1, x, 0);

  // ---- MoE feed-forward sub-block ----
  rmsnorm_kernel<<<NTOK, 256, 0, stream>>>(hbuf, fnw, gbf, gf32);

  gate_logits_kernel<<<(NTOK*EE + 255)/256, 256, 0, stream>>>(gf32, gw, logit);
  topk_kernel<<<(NTOK + 255)/256, 256, 0, stream>>>(logit, comb);

  // out starts as h; experts accumulate comb[m,e] * (hid @ w2[e])
  copy_kernel<<<(NTOK*CCH/4 + 255)/256, 256, 0, stream>>>(hbuf, out, NTOK*CCH/4);

  for (int e = 0; e < EE; ++e) {
    const float* w1e = w1 + (size_t)e * CCH * FFH;   // [C, FF]
    const float* w3e = w3 + (size_t)e * CCH * FFH;   // [C, FF]
    const float* w2e = w2 + (size_t)e * FFH * CCH;   // [FF, C]
    convT_kernel<<<dim3(CCH/32, FFH/32), cb, 0, stream>>>(w1e, weT1, CCH, FFH);
    convT_kernel<<<dim3(CCH/32, FFH/32), cb, 0, stream>>>(w3e, weT3, CCH, FFH);
    convT_kernel<<<dim3(FFH/32, CCH/32), cb, 0, stream>>>(w2e, weT2, FFH, CCH);

    wmma_gemm_bf16<<<dim3(NTOK/128, FFH/128), 256, 0, stream>>>(
        gbf, weT1, up1, NTOK, FFH, CCH, 0, nullptr, 0);
    wmma_gemm_bf16<<<dim3(NTOK/128, FFH/128), 256, 0, stream>>>(
        gbf, weT3, up3, NTOK, FFH, CCH, 0, nullptr, 0);

    silu_mul_kernel<<<(NTOK*FFH + 255)/256, 256, 0, stream>>>(up1, up3, hidb, NTOK*FFH);

    wmma_gemm_bf16<<<dim3(NTOK/128, CCH/128), 256, 0, stream>>>(
        hidb, weT2, out, NTOK, CCH, FFH, 2, comb + e, EE);
  }
}
